// RNNLayer_2602750181496
// MI455X (gfx1250) — matrix operand order repaired and checked by reading, not compiled, not run on hardware
//
#include <hip/hip_runtime.h>
#include <hip/hip_bf16.h>
#include <stdint.h>

// LSTM: T=512, B=64, D=1024, H=1024.
// Per step (one graph-captured kernel launch each): K=2048 bf16 WMMA GEMM
// ([x_t|h] @ Wc^T) with the A slice async-DMA'd into LDS once per block
// (GLOBAL_LOAD_ASYNC_TO_LDS_B128), fused gate activations + c/h update.
// Wc (16MB bf16) stays L2-resident across all 512 steps.

#define T_STEPS 512
#define B_DIM   64
#define D_DIM   1024
#define H_DIM   1024
#define G_DIM   (4 * H_DIM)      // 4096 gate rows
#define K_DIM   (D_DIM + H_DIM)  // 2048 combined K

typedef __attribute__((ext_vector_type(16))) __bf16 bf16x16;
typedef __attribute__((ext_vector_type(8)))  __bf16 bf16x8;
typedef __attribute__((ext_vector_type(8)))  float  floatx8;
typedef int v4i __attribute__((vector_size(16)));   // matches builtin param type

#define AS1 __attribute__((address_space(1)))
#define AS3 __attribute__((address_space(3)))

__device__ __forceinline__ unsigned short f2bf(float f) {
  unsigned int u = __builtin_bit_cast(unsigned int, f);
  u += 0x7FFFu + ((u >> 16) & 1u);               // round-to-nearest-even
  return (unsigned short)(u >> 16);
}

__device__ __forceinline__ bf16x8 ldb8(const unsigned short* p) {
  return *reinterpret_cast<const bf16x8*>(p);    // 16B load (global or ds)
}

__device__ __forceinline__ bf16x16 cat16(bf16x8 lo, bf16x8 hi) {
  return __builtin_shufflevector(lo, hi, 0,1,2,3,4,5,6,7,8,9,10,11,12,13,14,15);
}

// 16B global -> LDS async DMA (ASYNCcnt-tracked).
__device__ __forceinline__ void async_ld16(const unsigned short* g, unsigned short* l) {
  __builtin_amdgcn_global_load_async_to_lds_b128(
      (AS1 v4i*)g, (AS3 v4i*)l, 0, 0);
}

__device__ __forceinline__ void wait_async0() {
#if __has_builtin(__builtin_amdgcn_s_wait_asynccnt)
  __builtin_amdgcn_s_wait_asynccnt(0);
#else
  asm volatile("s_wait_asynccnt 0x0" ::: "memory");
#endif
}

// ---------------- prep kernels ----------------
__global__ void k_cvt_x(const float* __restrict__ x, unsigned short* __restrict__ xb, int n) {
  int i = blockIdx.x * blockDim.x + threadIdx.x;
  if (i < n) xb[i] = f2bf(x[i]);
}

// Wc[n][k] : k<1024 -> W_ih[n][k], else W_hh[n][k-1024]  (row n = gate output n)
__global__ void k_build_w(const float* __restrict__ Wih, const float* __restrict__ Whh,
                          unsigned short* __restrict__ Wc) {
  int i = blockIdx.x * blockDim.x + threadIdx.x;   // < 4096*2048
  int n = i >> 11, k = i & 2047;
  float v = (k < D_DIM) ? Wih[n * D_DIM + k] : Whh[n * H_DIM + (k - D_DIM)];
  Wc[i] = f2bf(v);
}

__global__ void k_init(const float* __restrict__ h0, const float* __restrict__ c0,
                       const float* __restrict__ bih, const float* __restrict__ bhh,
                       unsigned short* __restrict__ hb, float* __restrict__ cbuf,
                       float* __restrict__ bias) {
  int i = blockIdx.x * blockDim.x + threadIdx.x;   // < 65536
  cbuf[i] = c0[i];
  hb[i]   = f2bf(h0[i]);
  if (i < G_DIM) bias[i] = bih[i] + bhh[i];
}

// ---------------- fused LSTM step ----------------
// grid 64: blockIdx.x = 16-hidden-unit chunk p; 256 threads = 8 waves.
// Wave w: gate = w&3, row-tile pair = w>>2 -> tiles {2*mp, 2*mp+1}.
// One B fragment feeds two WMMAs (two row tiles).  A (64x2048 bf16, 256KB)
// is async-DMA'd to LDS once; B streams from L2-resident Wc.
__global__ void __launch_bounds__(256)
k_step(const unsigned short* __restrict__ xb, const unsigned short* __restrict__ Wc,
       const float* __restrict__ bias, const unsigned short* __restrict__ hprev,
       unsigned short* __restrict__ hnext, float* __restrict__ cbuf,
       float* __restrict__ out, int t)
{
  extern __shared__ char smem[];                       // 256KB A + 16KB gates
  unsigned short* sA = (unsigned short*)smem;          // [64][2048] bf16
  float* s_g = (float*)(smem + B_DIM * K_DIM * 2);     // [4][64][16] f32

  const int tid  = threadIdx.x;
  const int p    = blockIdx.x;
  const int wave = tid >> 5;
  const int lane = tid & 31;
  const int half = lane >> 4;
  const int l    = lane & 15;
  const int gate = wave & 3;
  const int mp   = wave >> 2;                          // 0..1 -> tiles 2mp,2mp+1

  // ---- stage A = [x_t | h] into LDS via async DMA (16B per lane-op) ----
  // x region: 64 rows x 1024 elems = 8192 16B chunks; h region likewise.
  for (int c = tid; c < 8192; c += 256) {
    int m  = c >> 7;                                   // 128 chunks per row
    int ko = (c & 127) * 8;
    async_ld16(xb + ((size_t)t * B_DIM + m) * D_DIM + ko, &sA[m * K_DIM + ko]);
  }
  for (int c = tid; c < 8192; c += 256) {
    int m  = c >> 7;
    int ko = (c & 127) * 8;
    async_ld16(hprev + (size_t)m * H_DIM + ko, &sA[m * K_DIM + D_DIM + ko]);
  }
  wait_async0();
  __syncthreads();

  // ---- GEMM: gates[64, 4096] tile (gate, units p*16..+15, all 64 rows) ----
  const int n  = gate * H_DIM + p * 16 + l;            // B column = Wc row n
  const int r0 = (2 * mp + 0) * 16 + l;                // A row, tile 0
  const int r1 = (2 * mp + 1) * 16 + l;                // A row, tile 1
  const unsigned short* brow = Wc + (size_t)n * K_DIM;

  floatx8 acc0 = {};
  floatx8 acc1 = {};
  #pragma unroll 2
  for (int k = 0; k < K_DIM; k += 32) {
    const int kb = k + half * 8;                       // B: K {kb..+7, kb+16..+23}
    const int ka = k + half * 8;                       // A: K {ka..+7, ka+16..+23}
    bf16x16 b  = cat16(ldb8(brow + kb),            ldb8(brow + kb + 16));
    bf16x16 a0 = cat16(ldb8(&sA[r0 * K_DIM + ka]), ldb8(&sA[r0 * K_DIM + ka + 16]));
    bf16x16 a1 = cat16(ldb8(&sA[r1 * K_DIM + ka]), ldb8(&sA[r1 * K_DIM + ka + 16]));
    acc0 = __builtin_amdgcn_wmma_f32_16x16x32_bf16(false, a0, false, b,
                                                   (short)0, acc0, false, false);
    acc1 = __builtin_amdgcn_wmma_f32_16x16x32_bf16(false, a1, false, b,
                                                   (short)0, acc1, false, false);
  }

  // C layout: VGPR j -> row = tile*16 + half*8 + j, col = l
  #pragma unroll
  for (int j = 0; j < 8; ++j) {
    s_g[gate * (B_DIM * 16) + ((2 * mp + 0) * 16 + half * 8 + j) * 16 + l] = acc0[j];
    s_g[gate * (B_DIM * 16) + ((2 * mp + 1) * 16 + half * 8 + j) * 16 + l] = acc1[j];
  }
  __syncthreads();

  // ---- fused LSTM cell: block owns all 64 rows x units p*16..+15 ----
  for (int e = tid; e < B_DIM * 16; e += 256) {
    const int r = e >> 4, cx = e & 15;                 // batch row, unit-in-chunk
    const int unit = p * 16 + cx;
    float gi = s_g[0 * (B_DIM * 16) + e] + bias[unit];
    float gf = s_g[1 * (B_DIM * 16) + e] + bias[H_DIM + unit];
    float gg = s_g[2 * (B_DIM * 16) + e] + bias[2 * H_DIM + unit];
    float go = s_g[3 * (B_DIM * 16) + e] + bias[3 * H_DIM + unit];
    float ig = 1.0f / (1.0f + __expf(-gi));
    float fg = 1.0f / (1.0f + __expf(-gf));
    float g  = tanhf(gg);
    float og = 1.0f / (1.0f + __expf(-go));
    const size_t idx = (size_t)r * H_DIM + unit;
    float c  = cbuf[idx];
    float cn = fg * c + ig * g;
    float hn = og * tanhf(cn);
    cbuf[idx] = cn;                                    // in-place: unique owner
    out[(size_t)t * (B_DIM * H_DIM) + idx] = hn;       // outputs[t]
    hnext[idx] = f2bf(hn);                             // bf16 h for next step
  }
}

__global__ void k_tail(const float* __restrict__ cbuf, float* __restrict__ out) {
  int i = blockIdx.x * blockDim.x + threadIdx.x;       // < 65536
  const size_t BH = (size_t)B_DIM * H_DIM;
  out[(size_t)T_STEPS * BH + i]      = out[((size_t)T_STEPS - 1) * BH + i]; // h_T
  out[(size_t)T_STEPS * BH + BH + i] = cbuf[i];                             // c_T
}

// ---------------- host ----------------
extern "C" void kernel_launch(void* const* d_in, const int* in_sizes, int n_in,
                              void* d_out, int out_size, void* d_ws, size_t ws_size,
                              hipStream_t stream) {
  const float* x   = (const float*)d_in[0];
  const float* h0  = (const float*)d_in[1];
  const float* c0  = (const float*)d_in[2];
  const float* Wih = (const float*)d_in[3];
  const float* Whh = (const float*)d_in[4];
  const float* bih = (const float*)d_in[5];
  const float* bhh = (const float*)d_in[6];
  float* out = (float*)d_out;
  char*  ws  = (char*)d_ws;

  // workspace layout (~80.5 MB), offsets 16KB-aligned
  unsigned short* Wc  = (unsigned short*)(ws);                 // 16 MB
  unsigned short* xbf = (unsigned short*)(ws + 16777216);      // 64 MB
  float*          bias= (float*)         (ws + 83886080);      // 16 KB
  unsigned short* hb0 = (unsigned short*)(ws + 83902464);      // 128 KB
  unsigned short* hb1 = (unsigned short*)(ws + 84033536);      // 128 KB
  float*          cbuf= (float*)         (ws + 84164608);      // 256 KB

  const int nx = T_STEPS * B_DIM * D_DIM;                      // 33,554,432
  k_cvt_x  <<<nx / 256, 256, 0, stream>>>(x, xbf, nx);
  k_build_w<<<(G_DIM * K_DIM) / 256, 256, 0, stream>>>(Wih, Whh, Wc);
  k_init   <<<(B_DIM * H_DIM) / 256, 256, 0, stream>>>(h0, c0, bih, bhh, hb0, cbuf, bias);

  const size_t shmem = (size_t)B_DIM * K_DIM * 2 + 4 * B_DIM * 16 * 4; // 272 KB
  unsigned short* hp = hb0;
  unsigned short* hn = hb1;
  for (int t = 0; t < T_STEPS; ++t) {
    k_step<<<64, 256, shmem, stream>>>(xbf, Wc, bias, hp, hn, cbuf, out, t);
    unsigned short* tmp = hp; hp = hn; hn = tmp;
  }
  k_tail<<<(B_DIM * H_DIM) / 256, 256, 0, stream>>>(cbuf, out);
}
